// MambaVisionMixer3d_60619168415820
// MI455X (gfx1250) — matrix operand order repaired and checked by reading, not compiled, not run on hardware
//
#include <hip/hip_runtime.h>
#include <hip/hip_bf16.h>
#include <cstdint>

// ---- problem dims ----
#define B_SZ 2
#define SEQ  4096
#define DM   512
#define DI   1024
#define E2   2048          // 2*DI
#define NST  16            // d_state
#define DTR  32            // dt_rank
#define NTOK (B_SZ*SEQ)    // 8192 tokens
#define SCH  16            // scan chunks
#define CHK  (SEQ/SCH)     // 256 steps per chunk

typedef __bf16 bf16_t;
typedef bf16_t v8bf  __attribute__((ext_vector_type(8)));
typedef bf16_t v16bf __attribute__((ext_vector_type(16)));
typedef float  v8f   __attribute__((ext_vector_type(8)));

static __device__ __forceinline__ v16bf cat16(v8bf lo, v8bf hi) {
  return __builtin_shufflevector(lo, hi, 0,1,2,3,4,5,6,7,8,9,10,11,12,13,14,15);
}

static __device__ __forceinline__ float softplus_f(float x) {
  return (x > 20.f) ? x : log1pf(__expf(x));
}

// ---------------- fp32 -> bf16 conversion ----------------
__global__ __launch_bounds__(256) void cvt_bf16_kernel(const float* __restrict__ src,
                                                       bf16_t* __restrict__ dst, size_t n) {
  size_t i = (size_t)blockIdx.x * 256 + threadIdx.x;
  if (i < n) dst[i] = (bf16_t)src[i];
}

// ---------------- generic bf16 WMMA GEMM: C[M,N] = A[M,K] * W[N,K]^T -------------
// One wave computes a 32(M) x 64(N) tile: 2 m-subtiles x 4 n-tiles = 8 accumulators.
// Software-pipelined: next k-step fragments are prefetched into a second register
// set before the current k-step's 8 WMMAs issue, so loads overlap matrix math.
// mode 0: store f32 C
// mode 1: store f32 C and bf16 copy Cb
// mode 2: store f32 softplus(C + bias[n])   (dt path -> delta)
__global__ __launch_bounds__(256) void gemm_bf16_wmma(
    const bf16_t* __restrict__ A, int lda,
    const bf16_t* __restrict__ Bw, int ldb,
    float* __restrict__ C, int ldc,
    bf16_t* __restrict__ Cb,
    const float* __restrict__ bias,
    int M, int N, int K, int mode)
{
  const int wave = (int)((blockIdx.x * blockDim.x + threadIdx.x) >> 5);
  const int lane = threadIdx.x & 31;
  const int nGroups = N >> 6;                 // tiles of 64 in N
  const int mT = wave / nGroups;              // 32-row tile index
  const int ng = wave % nGroups;
  if (mT * 32 >= M) return;                   // wave-uniform guard (EXEC stays full)

  const int h  = lane >> 4;                   // half-wave select
  const int mr = lane & 15;                   // row (A) / column (B,C)

  const bf16_t* Ap0 = A + (size_t)(mT * 32 + mr) * (size_t)lda;
  const bf16_t* Ap1 = Ap0 + (size_t)16 * (size_t)lda;
  const bf16_t* Bbase = Bw + (size_t)(ng * 64 + mr) * (size_t)ldb;
  const size_t bRow16 = (size_t)16 * (size_t)ldb;

  // A fragment (16x32 bf16): lane(h,mr) holds k in [kb+8h, kb+8h+8) U [kb+16+8h, ...)
  auto loadA = [&](const bf16_t* Ap, int kb) -> v16bf {
    v8bf lo = *(const v8bf*)(Ap + kb + 8 * h);
    v8bf hi = *(const v8bf*)(Ap + kb + 16 + 8 * h);
    return cat16(lo, hi);
  };
  // B fragment (32x16 bf16): lane(h,mr) holds k in [kb+8h, kb+8h+8) U [kb+16+8h, ...) at column mr
  auto loadB = [&](int j, int kb) -> v16bf {
    const bf16_t* Bp = Bbase + (size_t)j * bRow16 + kb + 8 * h;
    v8bf lo = *(const v8bf*)(Bp);
    v8bf hi = *(const v8bf*)(Bp + 16);
    return cat16(lo, hi);
  };

  const v8f vz = {0.f,0.f,0.f,0.f,0.f,0.f,0.f,0.f};
  v8f acc[8] = {vz, vz, vz, vz, vz, vz, vz, vz};

  // current fragment buffers
  v16bf a0c = loadA(Ap0, 0);
  v16bf a1c = loadA(Ap1, 0);
  v16bf bc0 = loadB(0, 0), bc1 = loadB(1, 0), bc2 = loadB(2, 0), bc3 = loadB(3, 0);

  for (int kb = 0; kb + 32 < K; kb += 32) {
    const int kn = kb + 32;
    // prefetch next k-step into second buffer set (issues before the WMMAs)
    v16bf a0n = loadA(Ap0, kn);
    v16bf a1n = loadA(Ap1, kn);
    v16bf bn0 = loadB(0, kn), bn1 = loadB(1, kn), bn2 = loadB(2, kn), bn3 = loadB(3, kn);
    // 8 back-to-back WMMAs on the current buffers (disjoint accumulators: no hazards)
    acc[0] = __builtin_amdgcn_wmma_f32_16x16x32_bf16(false, a0c, false, bc0, (short)0, acc[0], false, false);
    acc[4] = __builtin_amdgcn_wmma_f32_16x16x32_bf16(false, a1c, false, bc0, (short)0, acc[4], false, false);
    acc[1] = __builtin_amdgcn_wmma_f32_16x16x32_bf16(false, a0c, false, bc1, (short)0, acc[1], false, false);
    acc[5] = __builtin_amdgcn_wmma_f32_16x16x32_bf16(false, a1c, false, bc1, (short)0, acc[5], false, false);
    acc[2] = __builtin_amdgcn_wmma_f32_16x16x32_bf16(false, a0c, false, bc2, (short)0, acc[2], false, false);
    acc[6] = __builtin_amdgcn_wmma_f32_16x16x32_bf16(false, a1c, false, bc2, (short)0, acc[6], false, false);
    acc[3] = __builtin_amdgcn_wmma_f32_16x16x32_bf16(false, a0c, false, bc3, (short)0, acc[3], false, false);
    acc[7] = __builtin_amdgcn_wmma_f32_16x16x32_bf16(false, a1c, false, bc3, (short)0, acc[7], false, false);
    // rotate buffers
    a0c = a0n; a1c = a1n; bc0 = bn0; bc1 = bn1; bc2 = bn2; bc3 = bn3;
  }
  // final (peeled) k-step
  acc[0] = __builtin_amdgcn_wmma_f32_16x16x32_bf16(false, a0c, false, bc0, (short)0, acc[0], false, false);
  acc[4] = __builtin_amdgcn_wmma_f32_16x16x32_bf16(false, a1c, false, bc0, (short)0, acc[4], false, false);
  acc[1] = __builtin_amdgcn_wmma_f32_16x16x32_bf16(false, a0c, false, bc1, (short)0, acc[1], false, false);
  acc[5] = __builtin_amdgcn_wmma_f32_16x16x32_bf16(false, a1c, false, bc1, (short)0, acc[5], false, false);
  acc[2] = __builtin_amdgcn_wmma_f32_16x16x32_bf16(false, a0c, false, bc2, (short)0, acc[2], false, false);
  acc[6] = __builtin_amdgcn_wmma_f32_16x16x32_bf16(false, a1c, false, bc2, (short)0, acc[6], false, false);
  acc[3] = __builtin_amdgcn_wmma_f32_16x16x32_bf16(false, a0c, false, bc3, (short)0, acc[3], false, false);
  acc[7] = __builtin_amdgcn_wmma_f32_16x16x32_bf16(false, a1c, false, bc3, (short)0, acc[7], false, false);

  // C layout per 16x16 tile: VGPR r -> M = r + 8h ; N = lane & 15
#pragma unroll
  for (int ms = 0; ms < 2; ++ms) {
#pragma unroll
    for (int j = 0; j < 4; ++j) {
      const int n = ng * 64 + j * 16 + mr;
      const v8f& a = acc[ms * 4 + j];
#pragma unroll
      for (int r = 0; r < 8; ++r) {
        const int row = mT * 32 + ms * 16 + r + 8 * h;
        float v = a[r];
        if (mode == 2) v = softplus_f(v + bias[n]);
        C[(size_t)row * ldc + n] = v;
        if (mode == 1) Cb[(size_t)row * ldc + n] = (bf16_t)v;
      }
    }
  }
}

// ---------------- depthwise causal conv (k=4) + SiLU, token-major in/out ----------
// xz: (tok, 2048). c<1024 -> x path (writes xs f32 + xs bf16), else z path (zs f32).
__global__ __launch_bounds__(256) void conv_dw_silu(
    const float* __restrict__ xz,
    const float* __restrict__ wx, const float* __restrict__ bx,
    const float* __restrict__ wz, const float* __restrict__ bz,
    float* __restrict__ xs, bf16_t* __restrict__ xsb, float* __restrict__ zs)
{
  size_t idx = (size_t)blockIdx.x * 256 + threadIdx.x;
  const int c  = (int)(idx & (E2 - 1));       // channel fastest -> coalesced
  size_t t = idx >> 11;                       // / 2048
  const int l4 = (int)(t & (SEQ / 4 - 1));    // 4 outputs per thread along L
  const int b  = (int)(t >> 10);              // / (SEQ/4)
  if (b >= B_SZ) return;

  const bool isX = (c < DI);
  const int d = isX ? c : (c - DI);
  const float* w  = (isX ? wx : wz) + d * 4;
  const float  bc = (isX ? bx : bz)[d];
  const float w0 = w[0], w1 = w[1], w2 = w[2], w3 = w[3];

  const int lbase = l4 * 4;
  float in[7];
#pragma unroll
  for (int j = 0; j < 7; ++j) {
    const int l = lbase - 3 + j;
    in[j] = (l >= 0) ? xz[((size_t)b * SEQ + l) * E2 + c] : 0.f;
  }
#pragma unroll
  for (int i = 0; i < 4; ++i) {
    float a = bc + w0 * in[i] + w1 * in[i + 1] + w2 * in[i + 2] + w3 * in[i + 3];
    float s = a / (1.f + __expf(-a));         // SiLU
    const size_t off = ((size_t)b * SEQ + lbase + i) * DI + d;
    if (isX) { xs[off] = s; xsb[off] = (bf16_t)s; }
    else     { zs[off] = s; }
  }
}

// ---------------- chunked selective scan ----------------
// Pass 1: per (chunk s, b, d): local scan with h0=0, accumulate sum(delta).
__global__ __launch_bounds__(256) void scan_pass1(
    const float* __restrict__ delta, const float* __restrict__ xs,
    const float* __restrict__ xdbl,  const float* __restrict__ Alog,
    float* __restrict__ hloc, float* __restrict__ dsum)
{
  const int d = blockIdx.x * 256 + threadIdx.x;
  const int b = blockIdx.y;
  const int s = blockIdx.z;

  float Arow[NST];
#pragma unroll
  for (int n = 0; n < NST; ++n) Arow[n] = -__expf(Alog[d * NST + n]);

  float h[NST];
#pragma unroll
  for (int n = 0; n < NST; ++n) h[n] = 0.f;
  float ds = 0.f;

  const size_t tok0 = (size_t)b * SEQ + (size_t)s * CHK;
  for (int l = 0; l < CHK; ++l) {
    const size_t tok = tok0 + l;
    const float dt = delta[tok * DI + d];
    const float u  = xs[tok * DI + d];
    const float* Bc = xdbl + tok * 64 + DTR;  // wave-uniform address -> scalar loads
    const float du = dt * u;
#pragma unroll
    for (int n = 0; n < NST; ++n) {
      const float dA = __expf(dt * Arow[n]);
      h[n] = h[n] * dA + du * Bc[n];
    }
    ds += dt;
  }
  const size_t base = (((size_t)s * B_SZ + b) * DI + d);
#pragma unroll
  for (int n = 0; n < NST; ++n) hloc[base * NST + n] = h[n];
  dsum[base] = ds;
}

// Middle: per (b,d): sequential over chunks; chunk decay = exp(A * sum(delta)).
__global__ __launch_bounds__(256) void scan_mid(
    const float* __restrict__ hloc, const float* __restrict__ dsum,
    const float* __restrict__ Alog, float* __restrict__ hstart)
{
  const int idx = blockIdx.x * 256 + threadIdx.x;   // 0 .. B_SZ*DI-1
  const int b = idx / DI, d = idx % DI;

  float Arow[NST];
#pragma unroll
  for (int n = 0; n < NST; ++n) Arow[n] = -__expf(Alog[d * NST + n]);

  float h[NST];
#pragma unroll
  for (int n = 0; n < NST; ++n) h[n] = 0.f;

  for (int s = 0; s < SCH; ++s) {
    const size_t base = (((size_t)s * B_SZ + b) * DI + d);
#pragma unroll
    for (int n = 0; n < NST; ++n) hstart[base * NST + n] = h[n];
    const float ds = dsum[base];
#pragma unroll
    for (int n = 0; n < NST; ++n)
      h[n] = hloc[base * NST + n] + __expf(Arow[n] * ds) * h[n];
  }
}

// Pass 2: re-scan each chunk from its true h_start, emit gated output (bf16, token-major).
__global__ __launch_bounds__(256) void scan_pass2(
    const float* __restrict__ delta, const float* __restrict__ xs,
    const float* __restrict__ zs,    const float* __restrict__ xdbl,
    const float* __restrict__ Alog,  const float* __restrict__ Dp,
    const float* __restrict__ hstart, bf16_t* __restrict__ yg)
{
  const int d = blockIdx.x * 256 + threadIdx.x;
  const int b = blockIdx.y;
  const int s = blockIdx.z;

  float Arow[NST];
#pragma unroll
  for (int n = 0; n < NST; ++n) Arow[n] = -__expf(Alog[d * NST + n]);
  const float Dd = Dp[d];

  const size_t base = (((size_t)s * B_SZ + b) * DI + d);
  float h[NST];
#pragma unroll
  for (int n = 0; n < NST; ++n) h[n] = hstart[base * NST + n];

  const size_t tok0 = (size_t)b * SEQ + (size_t)s * CHK;
  for (int l = 0; l < CHK; ++l) {
    const size_t tok = tok0 + l;
    const float dt = delta[tok * DI + d];
    const float u  = xs[tok * DI + d];
    const float* Bc = xdbl + tok * 64 + DTR;        // uniform
    const float* Cc = Bc + NST;                     // uniform
    const float du = dt * u;
    float y = 0.f;
#pragma unroll
    for (int n = 0; n < NST; ++n) {
      const float dA = __expf(dt * Arow[n]);
      h[n] = h[n] * dA + du * Bc[n];
      y += h[n] * Cc[n];
    }
    y += u * Dd;
    const float zg = zs[tok * DI + d];
    yg[tok * DI + d] = (bf16_t)(y * zg);
  }
}

// ---------------- host side ----------------
extern "C" void kernel_launch(void* const* d_in, const int* in_sizes, int n_in,
                              void* d_out, int out_size, void* d_ws, size_t ws_size,
                              hipStream_t stream)
{
  (void)in_sizes; (void)n_in; (void)out_size; (void)ws_size;
  const float* hs   = (const float*)d_in[0];
  const float* Win  = (const float*)d_in[1];
  const float* cxw  = (const float*)d_in[2];
  const float* cxb  = (const float*)d_in[3];
  const float* czw  = (const float*)d_in[4];
  const float* czb  = (const float*)d_in[5];
  const float* Wxp  = (const float*)d_in[6];
  const float* Wdt  = (const float*)d_in[7];
  const float* bdt  = (const float*)d_in[8];
  const float* Alog = (const float*)d_in[9];
  const float* Dp   = (const float*)d_in[10];
  const float* Wout = (const float*)d_in[11];

  char* ws = (char*)d_ws; size_t off = 0;
  auto alloc = [&](size_t bytes) -> void* {
    off = (off + 255) & ~(size_t)255;
    void* p = ws + off; off += bytes; return p;
  };
  bf16_t* hs_b   = (bf16_t*)alloc((size_t)NTOK * DM  * 2);
  bf16_t* Win_b  = (bf16_t*)alloc((size_t)E2   * DM  * 2);
  bf16_t* Wxp_b  = (bf16_t*)alloc((size_t)64   * DI  * 2);
  bf16_t* Wdt_b  = (bf16_t*)alloc((size_t)DI   * DTR * 2);
  bf16_t* Wout_b = (bf16_t*)alloc((size_t)DM   * DI  * 2);
  float*  xz     = (float*) alloc((size_t)NTOK * E2  * 4);
  float*  xsf    = (float*) alloc((size_t)NTOK * DI  * 4);
  bf16_t* xsb    = (bf16_t*)alloc((size_t)NTOK * DI  * 2);
  float*  zsf    = (float*) alloc((size_t)NTOK * DI  * 4);
  float*  xdbl   = (float*) alloc((size_t)NTOK * 64  * 4);
  bf16_t* xdblb  = (bf16_t*)alloc((size_t)NTOK * 64  * 2);
  float*  deltaB = (float*) alloc((size_t)NTOK * DI  * 4);
  bf16_t* yg     = (bf16_t*)alloc((size_t)NTOK * DI  * 2);
  float*  hloc   = (float*) alloc((size_t)SCH * B_SZ * DI * NST * 4);
  float*  hstart = (float*) alloc((size_t)SCH * B_SZ * DI * NST * 4);
  float*  dsum   = (float*) alloc((size_t)SCH * B_SZ * DI * 4);

  auto cvt = [&](const float* s, bf16_t* dst, size_t n) {
    cvt_bf16_kernel<<<(unsigned)((n + 255) / 256), 256, 0, stream>>>(s, dst, n);
  };
  cvt(hs,   hs_b,   (size_t)NTOK * DM);
  cvt(Win,  Win_b,  (size_t)E2 * DM);
  cvt(Wxp,  Wxp_b,  (size_t)64 * DI);
  cvt(Wdt,  Wdt_b,  (size_t)DI * DTR);
  cvt(Wout, Wout_b, (size_t)DM * DI);

  // 1) in_proj: xz[tok, 2048] = hs[tok,512] * W_in^T
  {
    const int waves = (NTOK / 32) * (E2 / 64);
    gemm_bf16_wmma<<<waves / 8, 256, 0, stream>>>(hs_b, DM, Win_b, DM, xz, E2,
                                                  nullptr, nullptr, NTOK, E2, DM, 0);
  }
  // 2) depthwise causal conv + SiLU on both halves
  conv_dw_silu<<<(B_SZ * (SEQ / 4) * E2) / 256, 256, 0, stream>>>(
      xz, cxw, cxb, czw, czb, xsf, xsb, zsf);
  // 3) x_proj: x_dbl[tok,64] = xs * W_xproj^T   (f32 + bf16 copy)
  {
    const int waves = (NTOK / 32) * (64 / 64);
    gemm_bf16_wmma<<<waves / 8, 256, 0, stream>>>(xsb, DI, Wxp_b, DI, xdbl, 64,
                                                  xdblb, nullptr, NTOK, 64, DI, 1);
  }
  // 4) dt proj + softplus: delta[tok,1024] = softplus(dt_lo * W_dt^T + b_dt)
  {
    const int waves = (NTOK / 32) * (DI / 64);
    gemm_bf16_wmma<<<waves / 8, 256, 0, stream>>>(xdblb, 64, Wdt_b, DTR, deltaB, DI,
                                                  nullptr, bdt, NTOK, DI, DTR, 2);
  }
  // 5) chunked selective scan + D skip + z gating -> yg (bf16 token-major)
  scan_pass1<<<dim3(DI / 256, B_SZ, SCH), 256, 0, stream>>>(deltaB, xsf, xdbl, Alog, hloc, dsum);
  scan_mid  <<<(B_SZ * DI) / 256, 256, 0, stream>>>(hloc, dsum, Alog, hstart);
  scan_pass2<<<dim3(DI / 256, B_SZ, SCH), 256, 0, stream>>>(deltaB, xsf, zsf, xdbl, Alog, Dp,
                                                            hstart, yg);
  // 6) out_proj: out[tok,512] = yg * W_out^T
  {
    const int waves = (NTOK / 32) * (DM / 64);
    gemm_bf16_wmma<<<waves / 8, 256, 0, stream>>>(yg, DI, Wout_b, DI, (float*)d_out, DM,
                                                  nullptr, nullptr, NTOK, DM, DI, 0);
  }
}
